// DiffOp_8830452760922
// MI455X (gfx1250) — matrix operand order repaired and checked by reading, not compiled, not run on hardware
//
#include <hip/hip_runtime.h>
#include <hip/hip_bf16.h>

typedef __attribute__((ext_vector_type(16))) __bf16 v16bf;
typedef __attribute__((ext_vector_type(8)))  float  v8f;

#define N_INT   50000
#define N_BOUND 2000
#define E_INT   800000
#define E_BOUND 16000
#define T_STEPS 8
#define D_IN    64
#define D_EMB   128
#define D_MSG   128

// NOTE: E_INT, E_BOUND, N_INT, N_BOUND are all exact multiples of 16, so all
// 16-row tiles below are full tiles; no tail guards needed.

__device__ __forceinline__ unsigned short f2bf(float f) {
    unsigned int u = __builtin_bit_cast(unsigned int, f);
    u += 0x7FFFu + ((u >> 16) & 1u);      // round-to-nearest-even
    return (unsigned short)(u >> 16);
}

__device__ __forceinline__ float softplus_f(float x) {
    return (x > 20.0f) ? x : log1pf(expf(x));
}

// ---------------------------------------------------------------- utilities
__global__ void cvt_bf16_kernel(const float* __restrict__ src,
                                unsigned short* __restrict__ dst, int n) {
    int i = blockIdx.x * blockDim.x + threadIdx.x;
    if (i < n) dst[i] = f2bf(src[i]);
}

__global__ void bv_interp_kernel(const float* __restrict__ tptr,
                                 const float* __restrict__ ts,      // [T]
                                 const float* __restrict__ bvals,   // [T, N_BOUND, D_IN]
                                 float* __restrict__ out) {         // [N_BOUND, D_IN]
    int i = blockIdx.x * blockDim.x + threadIdx.x;
    if (i >= N_BOUND * D_IN) return;
    float t = tptr[0];
    int idx = 0;
    #pragma unroll
    for (int k = 0; k < T_STEPS; ++k) idx += (ts[k] < t) ? 1 : 0;  // searchsorted 'left'
    int il = idx - 1; if (il < 0) il = 0;
    int ir = idx;     if (ir > T_STEPS - 1) ir = T_STEPS - 1;
    float tl = ts[il], tr = ts[ir];
    float vl = bvals[(size_t)il * N_BOUND * D_IN + i];
    float vr = bvals[(size_t)ir * N_BOUND * D_IN + i];
    if (tl == tr) { out[i] = vl; }
    else          { out[i] = vl + (t - tl) * (vr - vl) / (tr - tl); }
}

__global__ void count_kernel(const int* __restrict__ dst, int nE,
                             float* __restrict__ cnt) {
    int e = blockIdx.x * blockDim.x + threadIdx.x;
    if (e < nE) atomicAdd(&cnt[dst[e]], 1.0f);
}

// ------------------------------------------------------------- msg GEMM
// raw_msg[e,:] = Wm * concat(src_feat[e], x[dst[e]]), scatter-add into seg[dst].
// (bias bm is mathematically hoisted into the node kernel: mean(m+b)=mean(m)+b)
template <int KDIM, bool SRC_BV>
__global__ void __launch_bounds__(128) msg_kernel(
    const float* __restrict__ xin,    // [*, KDIM/2] interior features
    const float* __restrict__ bvin,   // [N_BOUND, KDIM/2] (SRC_BV only)
    const int* __restrict__ srcIdx,
    const int* __restrict__ dstIdx,
    const unsigned short* __restrict__ Wm,  // [D_MSG, KDIM] bf16
    float* __restrict__ seg)                 // [N_INT, D_MSG]
{
    constexpr int D   = KDIM / 2;
    constexpr int LDK = KDIM + 4;            // pad 8B/row -> avoid bank conflicts
    __shared__ unsigned short tA[16 * LDK];
    __shared__ int sDst[16];

    const int tid  = threadIdx.x;
    const int tile = blockIdx.x * 16;

    { // ---- stage 16 gathered rows into LDS (bf16); all tiles are full
        const int r = tid >> 3, sub = tid & 7;
        const int e = tile + r;
        const int s = srcIdx[e];
        const int dnode = dstIdx[e];
        if (sub == 0) sDst[r] = dnode;
        const float* srow;
        if constexpr (SRC_BV) srow = bvin + (size_t)(s - N_INT) * D;
        else                  srow = xin + (size_t)s * D;
        const float* drow = xin + (size_t)dnode * D;
        constexpr int CH = KDIM / 8;
        #pragma unroll
        for (int c0 = 0; c0 < CH; ++c0) {
            int c = sub * CH + c0;
            float v = (c < D) ? srow[c] : drow[c - D];
            tA[r * LDK + c] = f2bf(v);
        }
    }
    __syncthreads();

    const int wave  = tid >> 5;
    const int lane  = tid & 31;
    const int col16 = lane & 15;
    const int hi    = (lane >= 16) ? 1 : 0;

    // hoist destination ids for this lane's 8 output rows
    int dvv[8];
    #pragma unroll
    for (int r = 0; r < 8; ++r) dvv[r] = sDst[r + (hi ? 8 : 0)];

    #pragma unroll
    for (int wt = 0; wt < 2; ++wt) {          // 8 N-tiles over 4 waves
        const int nbase = (wave * 2 + wt) * 16;
        v8f acc = {};
        #pragma unroll
        for (int k0 = 0; k0 < KDIM; k0 += 32) {
            union { unsigned int u[8]; v16bf v; } a, b;
            #pragma unroll
            for (int q = 0; q < 8; ++q) {
                // A 16x32 bf16 layout (ISA 7.12.2)
                int ka = k0 + 2 * (q & 3) + ((q >= 4) ? 16 : 0) + (hi ? 8 : 0);
                a.u[q] = *reinterpret_cast<const unsigned int*>(&tA[col16 * LDK + ka]);
                // B 32x16 bf16: same K order as A
                int kb = k0 + 2 * (q & 3) + ((q >= 4) ? 16 : 0) + (hi ? 8 : 0);
                b.u[q] = *reinterpret_cast<const unsigned int*>(
                    &Wm[(size_t)(nbase + col16) * KDIM + kb]);
            }
            acc = __builtin_amdgcn_wmma_f32_16x16x32_bf16(
                false, a.v, false, b.v, (short)0, acc, false, false);
        }
        const int col = nbase + col16;
        #pragma unroll
        for (int r = 0; r < 8; ++r) {          // C/D: VGPR r -> M = r (+8 for hi lanes)
            atomicAdd(&seg[(size_t)dvv[r] * D_MSG + col], acc[r]);
        }
    }
}

// ------------------------------------------------------------- node update
// agg = seg/max(cnt,1) + (cnt>0 ? bm : 0);  out = Ws*x + Wu*agg + bs + bu (+softplus)
template <int DK1, int DOUT, bool ACT>
__global__ void __launch_bounds__(128) node_kernel(
    const float* __restrict__ xin,   // [N, DK1]
    const float* __restrict__ seg,   // [N, 128]
    const float* __restrict__ cnt,   // [N]
    const float* __restrict__ bm,    // [128] message bias (hoisted)
    const unsigned short* __restrict__ Ws,  // [DOUT, DK1] bf16
    const float* __restrict__ bs,
    const unsigned short* __restrict__ Wu,  // [DOUT, 128] bf16
    const float* __restrict__ bu,
    float* __restrict__ xout)
{
    constexpr int L1 = DK1 + 4;
    constexpr int L2 = 128 + 4;
    __shared__ unsigned short t1[16 * L1];
    __shared__ unsigned short t2[16 * L2];

    const int tid  = threadIdx.x;
    const int tile = blockIdx.x * 16;
    {
        const int r = tid >> 3, sub = tid & 7;
        const int n = tile + r;
        const float cv  = cnt[n];
        const float inv = 1.0f / fmaxf(cv, 1.0f);
        const float bsc = (cv > 0.0f) ? 1.0f : 0.0f;
        constexpr int C1 = DK1 / 8;
        #pragma unroll
        for (int c0 = 0; c0 < C1; ++c0) {
            int c = sub * C1 + c0;
            t1[r * L1 + c] = f2bf(xin[(size_t)n * DK1 + c]);
        }
        #pragma unroll
        for (int c0 = 0; c0 < 16; ++c0) {
            int c = sub * 16 + c0;
            t2[r * L2 + c] = f2bf(seg[(size_t)n * 128 + c] * inv + bsc * bm[c]);
        }
    }
    __syncthreads();

    const int wave  = tid >> 5;
    const int lane  = tid & 31;
    const int col16 = lane & 15;
    const int hi    = (lane >= 16) ? 1 : 0;
    constexpr int NT = DOUT / 16;

    for (int nt = wave; nt < NT; nt += 4) {
        const int nbase = nt * 16;
        v8f acc = {};
        #pragma unroll
        for (int k0 = 0; k0 < DK1; k0 += 32) {
            union { unsigned int u[8]; v16bf v; } a, b;
            #pragma unroll
            for (int q = 0; q < 8; ++q) {
                int ka = k0 + 2 * (q & 3) + ((q >= 4) ? 16 : 0) + (hi ? 8 : 0);
                a.u[q] = *reinterpret_cast<const unsigned int*>(&t1[col16 * L1 + ka]);
                int kb = k0 + 2 * (q & 3) + ((q >= 4) ? 16 : 0) + (hi ? 8 : 0);
                b.u[q] = *reinterpret_cast<const unsigned int*>(
                    &Ws[(size_t)(nbase + col16) * DK1 + kb]);
            }
            acc = __builtin_amdgcn_wmma_f32_16x16x32_bf16(
                false, a.v, false, b.v, (short)0, acc, false, false);
        }
        #pragma unroll
        for (int k0 = 0; k0 < 128; k0 += 32) {
            union { unsigned int u[8]; v16bf v; } a, b;
            #pragma unroll
            for (int q = 0; q < 8; ++q) {
                int ka = k0 + 2 * (q & 3) + ((q >= 4) ? 16 : 0) + (hi ? 8 : 0);
                a.u[q] = *reinterpret_cast<const unsigned int*>(&t2[col16 * L2 + ka]);
                int kb = k0 + 2 * (q & 3) + ((q >= 4) ? 16 : 0) + (hi ? 8 : 0);
                b.u[q] = *reinterpret_cast<const unsigned int*>(
                    &Wu[(size_t)(nbase + col16) * 128 + kb]);
            }
            acc = __builtin_amdgcn_wmma_f32_16x16x32_bf16(
                false, a.v, false, b.v, (short)0, acc, false, false);
        }
        const int col = nbase + col16;
        const float bias = bs[col] + bu[col];
        #pragma unroll
        for (int r = 0; r < 8; ++r) {
            int nrow = tile + r + (hi ? 8 : 0);
            float val = acc[r] + bias;
            if (ACT) val = softplus_f(val);
            xout[(size_t)nrow * DOUT + col] = val;
        }
    }
}

// ------------------------------------------------------------- boundary update
template <int DK, int DOUT, bool ACT>
__global__ void __launch_bounds__(128) bv_kernel(
    const float* __restrict__ xin,          // [N, DK]
    const unsigned short* __restrict__ Ws,  // [DOUT, DK] bf16
    const float* __restrict__ bs,
    float* __restrict__ xout)
{
    constexpr int L1 = DK + 4;
    __shared__ unsigned short t1[16 * L1];

    const int tid  = threadIdx.x;
    const int tile = blockIdx.x * 16;
    {
        const int r = tid >> 3, sub = tid & 7;
        const int n = tile + r;
        constexpr int C1 = DK / 8;
        #pragma unroll
        for (int c0 = 0; c0 < C1; ++c0) {
            int c = sub * C1 + c0;
            t1[r * L1 + c] = f2bf(xin[(size_t)n * DK + c]);
        }
    }
    __syncthreads();

    const int wave  = tid >> 5;
    const int lane  = tid & 31;
    const int col16 = lane & 15;
    const int hi    = (lane >= 16) ? 1 : 0;
    constexpr int NT = DOUT / 16;

    for (int nt = wave; nt < NT; nt += 4) {
        const int nbase = nt * 16;
        v8f acc = {};
        #pragma unroll
        for (int k0 = 0; k0 < DK; k0 += 32) {
            union { unsigned int u[8]; v16bf v; } a, b;
            #pragma unroll
            for (int q = 0; q < 8; ++q) {
                int ka = k0 + 2 * (q & 3) + ((q >= 4) ? 16 : 0) + (hi ? 8 : 0);
                a.u[q] = *reinterpret_cast<const unsigned int*>(&t1[col16 * L1 + ka]);
                int kb = k0 + 2 * (q & 3) + ((q >= 4) ? 16 : 0) + (hi ? 8 : 0);
                b.u[q] = *reinterpret_cast<const unsigned int*>(
                    &Ws[(size_t)(nbase + col16) * DK + kb]);
            }
            acc = __builtin_amdgcn_wmma_f32_16x16x32_bf16(
                false, a.v, false, b.v, (short)0, acc, false, false);
        }
        const int col = nbase + col16;
        const float bias = bs[col];
        #pragma unroll
        for (int r = 0; r < 8; ++r) {
            int nrow = tile + r + (hi ? 8 : 0);
            float val = acc[r] + bias;
            if (ACT) val = softplus_f(val);
            xout[(size_t)nrow * DOUT + col] = val;
        }
    }
}

// ---------------------------------------------------------------- launcher
extern "C" void kernel_launch(void* const* d_in, const int* in_sizes, int n_in,
                              void* d_out, int out_size, void* d_ws, size_t ws_size,
                              hipStream_t stream) {
    (void)in_sizes; (void)n_in; (void)out_size; (void)ws_size;

    const float* tS    = (const float*)d_in[0];
    const float* x0    = (const float*)d_in[1];   // [N_INT, 64]
    const float* bvals = (const float*)d_in[2];   // [T, 1, N_BOUND, 64]
    const int*   ei    = (const int*)d_in[3];     // [2, E_INT]
    const int*   eb    = (const int*)d_in[4];     // [2, E_BOUND]
    const float* ts    = (const float*)d_in[5];   // [T]

    const float* Wf[4][3]; const float* Bf[4][3];
    for (int l = 0; l < 4; ++l)
        for (int j = 0; j < 3; ++j) {
            Wf[l][j] = (const float*)d_in[6 + l * 6 + j * 2];
            Bf[l][j] = (const float*)d_in[6 + l * 6 + j * 2 + 1];
        }

    // ---- carve workspace
    char* p = (char*)d_ws;
    auto carve = [&](size_t bytes) -> void* {
        void* r = (void*)p;
        p += (bytes + 255) & ~(size_t)255;
        return r;
    };
    float* xA  = (float*)carve((size_t)N_INT * 128 * 4);
    float* xB  = (float*)carve((size_t)N_INT * 128 * 4);
    float* seg = (float*)carve((size_t)N_INT * 128 * 4);
    float* cnt = (float*)carve((size_t)N_INT * 4);
    float* bvA = (float*)carve((size_t)N_BOUND * 128 * 4);
    float* bvB = (float*)carve((size_t)N_BOUND * 128 * 4);

    static const int wElems[4][3] = {
        {128 * 128, 128 * 64, 128 * 128},   // layer1: Wm, Ws, Wu
        {128 * 256, 128 * 128, 128 * 128},
        {128 * 256, 128 * 128, 128 * 128},
        {128 * 256, 64 * 128, 64 * 128}};
    unsigned short* Wbf[4][3];
    for (int l = 0; l < 4; ++l)
        for (int j = 0; j < 3; ++j) {
            int n = wElems[l][j];
            Wbf[l][j] = (unsigned short*)carve((size_t)n * 2);
            cvt_bf16_kernel<<<(n + 255) / 256, 256, 0, stream>>>(Wf[l][j], Wbf[l][j], n);
        }

    // ---- boundary interpolation
    bv_interp_kernel<<<(N_BOUND * D_IN + 255) / 256, 256, 0, stream>>>(tS, ts, bvals, bvA);

    // ---- degree counts (layer-invariant)
    hipMemsetAsync(cnt, 0, (size_t)N_INT * 4, stream);
    count_kernel<<<(E_INT + 255) / 256, 256, 0, stream>>>(ei + E_INT, E_INT, cnt);
    count_kernel<<<(E_BOUND + 255) / 256, 256, 0, stream>>>(eb + E_BOUND, E_BOUND, cnt);

    const int mtI = E_INT / 16;      // 50000 (exact)
    const int mtB = E_BOUND / 16;    // 1000  (exact)
    const int ntN = N_INT / 16;      // 3125  (exact)
    const int ntB = N_BOUND / 16;    // 125   (exact)
    const size_t segB = (size_t)N_INT * 128 * 4;

    // ---------------- layer 1 (d=64 -> 128)
    hipMemsetAsync(seg, 0, segB, stream);
    msg_kernel<128, false><<<mtI, 128, 0, stream>>>(x0, nullptr, ei, ei + E_INT,
                                                    Wbf[0][0], seg);
    msg_kernel<128, true><<<mtB, 128, 0, stream>>>(x0, bvA, eb, eb + E_BOUND,
                                                   Wbf[0][0], seg);
    node_kernel<64, 128, true><<<ntN, 128, 0, stream>>>(x0, seg, cnt, Bf[0][0],
                                                        Wbf[0][1], Bf[0][1],
                                                        Wbf[0][2], Bf[0][2], xA);
    bv_kernel<64, 128, true><<<ntB, 128, 0, stream>>>(bvA, Wbf[0][1], Bf[0][1], bvB);

    // ---------------- layer 2 (128 -> 128)
    hipMemsetAsync(seg, 0, segB, stream);
    msg_kernel<256, false><<<mtI, 128, 0, stream>>>(xA, nullptr, ei, ei + E_INT,
                                                    Wbf[1][0], seg);
    msg_kernel<256, true><<<mtB, 128, 0, stream>>>(xA, bvB, eb, eb + E_BOUND,
                                                   Wbf[1][0], seg);
    node_kernel<128, 128, true><<<ntN, 128, 0, stream>>>(xA, seg, cnt, Bf[1][0],
                                                         Wbf[1][1], Bf[1][1],
                                                         Wbf[1][2], Bf[1][2], xB);
    bv_kernel<128, 128, true><<<ntB, 128, 0, stream>>>(bvB, Wbf[1][1], Bf[1][1], bvA);

    // ---------------- layer 3 (128 -> 128)
    hipMemsetAsync(seg, 0, segB, stream);
    msg_kernel<256, false><<<mtI, 128, 0, stream>>>(xB, nullptr, ei, ei + E_INT,
                                                    Wbf[2][0], seg);
    msg_kernel<256, true><<<mtB, 128, 0, stream>>>(xB, bvA, eb, eb + E_BOUND,
                                                   Wbf[2][0], seg);
    node_kernel<128, 128, true><<<ntN, 128, 0, stream>>>(xB, seg, cnt, Bf[2][0],
                                                         Wbf[2][1], Bf[2][1],
                                                         Wbf[2][2], Bf[2][2], xA);
    bv_kernel<128, 128, true><<<ntB, 128, 0, stream>>>(bvA, Wbf[2][1], Bf[2][1], bvB);

    // ---------------- layer 4 (128 -> 64), no activation, writes d_out
    hipMemsetAsync(seg, 0, segB, stream);
    msg_kernel<256, false><<<mtI, 128, 0, stream>>>(xA, nullptr, ei, ei + E_INT,
                                                    Wbf[3][0], seg);
    msg_kernel<256, true><<<mtB, 128, 0, stream>>>(xA, bvB, eb, eb + E_BOUND,
                                                   Wbf[3][0], seg);
    node_kernel<128, 64, false><<<ntN, 128, 0, stream>>>(xA, seg, cnt, Bf[3][0],
                                                         Wbf[3][1], Bf[3][1],
                                                         Wbf[3][2], Bf[3][2],
                                                         (float*)d_out);
}
